// GuidedSparseAttention_82265803588064
// MI455X (gfx1250) — matrix operand order repaired and checked by reading, not compiled, not run on hardware
//
#include <hip/hip_runtime.h>
#include <hip/hip_bf16.h>

#define DIMC  256
#define NSEQ  4096
#define NB    2
#define NHEAD 8
#define HD    32
#define KLEN  32
#define NLOW  256
#define BM    64
#define LDA   (DIMC + 8)   // half elements; row = 528 B (16B aligned)
#define KPAD  40           // 32 k + 8 pad; row = 80 B (16B aligned)

typedef __attribute__((ext_vector_type(16))) _Float16 v16h;
typedef __attribute__((ext_vector_type(8)))  _Float16 v8h;
typedef __attribute__((ext_vector_type(8)))  float    v8f;

union frag16 { v16h v; v8h h[2]; };

__device__ __forceinline__ v8f wmma16(v16h a, v16h b, v8f c) {
  return __builtin_amdgcn_wmma_f32_16x16x32_f16(false, a, false, b, (short)0, c,
                                                false, false);
}

// A fragment: logical 16x32 f16, storage row-major base[m*stride + k].
// Per lane: two contiguous 8-half runs -> two b128 LDS loads.
__device__ __forceinline__ v16h frag_a(const _Float16* base, int stride, int lane) {
  const int m = lane & 15, half = lane >> 4;
  const _Float16* p = base + m * stride + half * 8;
  frag16 f;
  f.h[0] = *(const v8h*)(p);        // K = half*8 .. +7      (v0..v3)
  f.h[1] = *(const v8h*)(p + 16);   // K = 16+half*8 .. +7   (v4..v7)
  return f.v;
}

// B fragment: logical 32x16 f16 B[k][n], storage column-major baseT[n*stride + k].
// Per lane: two contiguous 8-half runs -> two b128 LDS loads.
__device__ __forceinline__ v16h frag_bk(const _Float16* baseT, int stride, int lane) {
  const int n = lane & 15, grp = lane >> 4;
  const _Float16* p = baseT + n * stride + grp * 8;
  frag16 f;
  f.h[0] = *(const v8h*)(p);        // K = grp*8 .. +7
  f.h[1] = *(const v8h*)(p + 16);   // K = 16+grp*8 .. +7
  return f.v;
}

// -------- top-2 coarse blocks -> 32 high-res key indices per group --------
__global__ void k_topk(const float* __restrict__ cmap, int* __restrict__ idx) {
  int t = blockIdx.x * blockDim.x + threadIdx.x;
  if (t >= NB * NLOW) return;
  const float* row = cmap + (size_t)t * NLOW;
  float v1 = -3.4e38f, v2 = -3.4e38f;
  int   i1 = 0, i2 = 0;
  for (int j = 0; j < NLOW; ++j) {
    float v = row[j];
    if (v > v1)      { v2 = v1; i2 = i1; v1 = v; i1 = j; }
    else if (v > v2) { v2 = v; i2 = j; }
  }
  int tk[2] = {i1, i2};
#pragma unroll
  for (int s = 0; s < 2; ++s) {
    int r0 = (tk[s] >> 4) << 2, c0 = (tk[s] & 15) << 2;
#pragma unroll
    for (int d = 0; d < 16; ++d)
      idx[t * KLEN + s * 16 + d] = (r0 + (d >> 2)) * 64 + (c0 + (d & 3));
  }
}

// -------- projection: out[b,n,:] = X[b,:,n]·W + bias   (f16 output) --------
__global__ void __launch_bounds__(256)
k_proj(const float* __restrict__ X, const float* __restrict__ W,
       const float* __restrict__ bias, _Float16* __restrict__ outh) {
  __shared__ alignas(16) _Float16 sA[BM][LDA];
  __shared__ alignas(16) _Float16 sWt[DIMC][KPAD];   // W slab, transposed [nout][k]
  const int tid = threadIdx.x, lane = tid & 31, w = tid >> 5;
  const int b  = blockIdx.x / (NSEQ / BM);
  const int n0 = (blockIdx.x % (NSEQ / BM)) * BM;

  // stage A: sA[m][k] = X[b, k, n0+m]  (coalesced along m)
  const float* Xb = X + (size_t)b * DIMC * NSEQ;
#pragma unroll 4
  for (int it = 0; it < BM * DIMC / 256; ++it) {
    int e = it * 256 + tid;
    int m = e & (BM - 1);
    int k = e >> 6;
    sA[m][k] = (_Float16)Xb[(size_t)k * NSEQ + n0 + m];
  }

  const int rt = w & 3;   // 4 row tiles of 16
  const int nh = w >> 2;  // 2 column halves of 128
  v8f acc[8];
#pragma unroll
  for (int i = 0; i < 8; ++i) acc[i] = v8f{};

  for (int kk = 0; kk < DIMC / 32; ++kk) {
    __syncthreads();
    // stage W slab rows kk*32..+31, transposed into sWt[j][i]
#pragma unroll 4
    for (int it = 0; it < 32 * DIMC / 256; ++it) {
      int e = it * 256 + tid;
      int j = e & (DIMC - 1);
      int i = e >> 8;
      sWt[j][i] = (_Float16)W[(size_t)(kk * 32 + i) * DIMC + j];
    }
    __syncthreads();
    v16h a = frag_a(&sA[rt * 16][kk * 32], LDA, lane);
#pragma unroll
    for (int nt = 0; nt < 8; ++nt) {
      v16h bf = frag_bk(&sWt[nh * 128 + nt * 16][0], KPAD, lane);
      acc[nt] = wmma16(a, bf, acc[nt]);
    }
  }

  // stage output tile into sA (reuse), then 128-bit coalesced global stores
  __syncthreads();
  const int mbase = rt * 16 + 8 * (lane >> 4);
#pragma unroll
  for (int nt = 0; nt < 8; ++nt) {
    int nout = nh * 128 + nt * 16 + (lane & 15);
    float bi = bias[nout];
#pragma unroll
    for (int r = 0; r < 8; ++r)
      sA[mbase + r][nout] = (_Float16)(acc[nt][r] + bi);
  }
  __syncthreads();
  _Float16* outb = outh + ((size_t)b * NSEQ + n0) * DIMC;
#pragma unroll
  for (int it = 0; it < 8; ++it) {
    int ch = it * 256 + tid;
    int row = ch >> 5, colb = (ch & 31) * 8;
    *(v8h*)(outb + (size_t)row * DIMC + colb) = *(const v8h*)&sA[row][colb];
  }
}

// -------- sparse attention: one block per (batch, query-group of 16) --------
__global__ void __launch_bounds__(256)
k_attn(const _Float16* __restrict__ qh, const _Float16* __restrict__ kh,
       const _Float16* __restrict__ vh, const int* __restrict__ idx,
       _Float16* __restrict__ xh) {
  __shared__ alignas(16) _Float16 sQ[16][LDA];          // also reused for output
  __shared__ alignas(16) _Float16 sK[KLEN][LDA];        // [key][channel]
  __shared__ alignas(16) _Float16 sVt[DIMC][KPAD];      // [channel][key]
  __shared__ alignas(16) _Float16 sP[NHEAD][16][KPAD];  // probs [m][key]

  const int tid = threadIdx.x, lane = tid & 31, w = tid >> 5;
  const int b = blockIdx.x / NLOW, g = blockIdx.x % NLOW;
  const int n0 = g * 16;

  const _Float16* qb = qh + ((size_t)b * NSEQ + n0) * DIMC;
  const int* ib = idx + (b * NLOW + g) * KLEN;
  const _Float16* kb = kh + (size_t)b * NSEQ * DIMC;
  const _Float16* vb = vh + (size_t)b * NSEQ * DIMC;

#pragma unroll
  for (int it = 0; it < 2; ++it) {               // sQ: 16 rows x 256
    int ch = it * 256 + tid;
    int row = ch >> 5, colb = (ch & 31) * 8;
    *(v8h*)&sQ[row][colb] = *(const v8h*)(qb + (size_t)row * DIMC + colb);
  }
#pragma unroll
  for (int it = 0; it < 4; ++it) {               // sK: 32 gathered rows
    int ch = it * 256 + tid;
    int row = ch >> 5, colb = (ch & 31) * 8;
    int ki = ib[row];
    *(v8h*)&sK[row][colb] = *(const v8h*)(kb + (size_t)ki * DIMC + colb);
  }
#pragma unroll
  for (int it = 0; it < 4; ++it) {               // sVt: gathered, transposed
    int ch = it * 256 + tid;
    int row = ch >> 5, colb = (ch & 31) * 8;
    int ki = ib[row];
    v8h gv = *(const v8h*)(vb + (size_t)ki * DIMC + colb);
#pragma unroll
    for (int j = 0; j < 8; ++j) sVt[colb + j][row] = gv[j];
  }
  __syncthreads();

  const int h = w, c0 = h * HD;
  const float scale = 0.17677669529663687f;  // 1/sqrt(32)
  const int key = lane & 15, grp = lane >> 4;

  v8f zero = v8f{};
  v16h aq = frag_a(&sQ[0][c0], LDA, lane);
  // scores: B[k=channel][n=key] stored as sK[key][channel] -> column-major
  v8f s0 = wmma16(aq, frag_bk(&sK[0][c0],  LDA, lane), zero);   // keys 0..15
  v8f s1 = wmma16(aq, frag_bk(&sK[16][c0], LDA, lane), zero);   // keys 16..31

#pragma unroll
  for (int r = 0; r < 8; ++r) {
    float x0 = s0[r] * scale, x1 = s1[r] * scale;
    float mx = fmaxf(x0, x1);
#pragma unroll
    for (int off = 1; off < 16; off <<= 1)
      mx = fmaxf(mx, __shfl_xor(mx, off, 32));
    float e0 = __expf(x0 - mx), e1 = __expf(x1 - mx);
    float sm = e0 + e1;
#pragma unroll
    for (int off = 1; off < 16; off <<= 1)
      sm += __shfl_xor(sm, off, 32);
    float inv = 1.f / sm;
    int m = r + 8 * grp;                    // C-layout row for this VGPR/lane
    sP[h][m][key]      = (_Float16)(e0 * inv);
    sP[h][m][key + 16] = (_Float16)(e1 * inv);
  }
  __syncthreads();

  v16h ap = frag_a(&sP[h][0][0], KPAD, lane);
  // x: B[k=key][n=dim] stored as sVt[channel][key] -> column-major
  v8f o0 = wmma16(ap, frag_bk(&sVt[c0][0],      KPAD, lane), zero);  // dims 0..15
  v8f o1 = wmma16(ap, frag_bk(&sVt[c0 + 16][0], KPAD, lane), zero);  // dims 16..31

  // stage result into sQ (reuse), then 128-bit coalesced stores
#pragma unroll
  for (int r = 0; r < 8; ++r) {
    int m = r + 8 * grp;
    sQ[m][c0 + key]      = (_Float16)o0[r];
    sQ[m][c0 + 16 + key] = (_Float16)o1[r];
  }
  __syncthreads();
  _Float16* xo = xh + ((size_t)b * NSEQ + n0) * DIMC;
#pragma unroll
  for (int it = 0; it < 2; ++it) {
    int ch = it * 256 + tid;
    int row = ch >> 5, colb = (ch & 31) * 8;
    *(v8h*)(xo + (size_t)row * DIMC + colb) = *(const v8h*)&sQ[row][colb];
  }
}

// -------- output projection, computed transposed --------
// out[b,c,n] = sum_k W[k,c] * x[b,n,k] + bo[c]
// A = W^T (M = c), B = x^T (N = n) -> C-tile store is contiguous along n.
__global__ void __launch_bounds__(256)
k_outproj(const _Float16* __restrict__ Xh, const float* __restrict__ W,
          const float* __restrict__ bias, float* __restrict__ out) {
  __shared__ alignas(16) _Float16 sX[BM][LDA];        // x rows [n][k]
  __shared__ alignas(16) _Float16 sWt[DIMC][KPAD];    // W slab transposed [c][k]
  const int tid = threadIdx.x, lane = tid & 31, w = tid >> 5;
  const int b  = blockIdx.x / (NSEQ / BM);
  const int n0 = (blockIdx.x % (NSEQ / BM)) * BM;

  const _Float16* Xb = Xh + ((size_t)b * NSEQ + n0) * DIMC;
#pragma unroll 4
  for (int it = 0; it < 8; ++it) {               // 64 rows x 256, v8h chunks
    int ch = it * 256 + tid;
    int row = ch >> 5, colb = (ch & 31) * 8;
    *(v8h*)&sX[row][colb] = *(const v8h*)(Xb + (size_t)row * DIMC + colb);
  }

  // wave w owns m-tiles (c) {2w, 2w+1} and all 4 n-tiles
  v8f acc[2][4];
#pragma unroll
  for (int i = 0; i < 2; ++i)
#pragma unroll
    for (int j = 0; j < 4; ++j) acc[i][j] = v8f{};

  for (int kk = 0; kk < DIMC / 32; ++kk) {
    __syncthreads();
#pragma unroll 4
    for (int it = 0; it < 32 * DIMC / 256; ++it) {
      int e = it * 256 + tid;
      int j = e & (DIMC - 1);
      int i = e >> 8;
      sWt[j][i] = (_Float16)W[(size_t)(kk * 32 + i) * DIMC + j];
    }
    __syncthreads();
#pragma unroll
    for (int mtl = 0; mtl < 2; ++mtl) {
      v16h a = frag_a(&sWt[(w * 2 + mtl) * 16][0], KPAD, lane);
#pragma unroll
      for (int nt = 0; nt < 4; ++nt) {
        v16h bf = frag_bk(&sX[nt * 16][kk * 32], LDA, lane);
        acc[mtl][nt] = wmma16(a, bf, acc[mtl][nt]);
      }
    }
  }

  const int grp = lane >> 4;
  float* outb = out + (size_t)b * DIMC * NSEQ + n0;
#pragma unroll
  for (int mtl = 0; mtl < 2; ++mtl) {
#pragma unroll
    for (int r = 0; r < 8; ++r) {
      int c = (w * 2 + mtl) * 16 + r + 8 * grp;
      float bi = bias[c];
#pragma unroll
      for (int nt = 0; nt < 4; ++nt)
        outb[(size_t)c * NSEQ + nt * 16 + (lane & 15)] = acc[mtl][nt][r] + bi;
    }
  }
}

extern "C" void kernel_launch(void* const* d_in, const int* in_sizes, int n_in,
                              void* d_out, int out_size, void* d_ws, size_t ws_size,
                              hipStream_t stream) {
  (void)in_sizes; (void)n_in; (void)out_size; (void)ws_size;
  const float* qf   = (const float*)d_in[0];
  const float* kf   = (const float*)d_in[1];
  const float* vf   = (const float*)d_in[2];
  const float* cmap = (const float*)d_in[3];
  const float* Wq   = (const float*)d_in[4];
  const float* bq   = (const float*)d_in[5];
  const float* Wk   = (const float*)d_in[6];
  const float* bk   = (const float*)d_in[7];
  const float* Wv   = (const float*)d_in[8];
  const float* bv   = (const float*)d_in[9];
  const float* Wo   = (const float*)d_in[10];
  const float* bo   = (const float*)d_in[11];
  float* out = (float*)d_out;

  char* ws = (char*)d_ws;
  const size_t projBytes = (size_t)NB * NSEQ * DIMC * sizeof(_Float16);
  _Float16* qhp = (_Float16*)ws; ws += projBytes;
  _Float16* khp = (_Float16*)ws; ws += projBytes;
  _Float16* vhp = (_Float16*)ws; ws += projBytes;
  _Float16* xh  = (_Float16*)ws; ws += projBytes;
  int* idx = (int*)ws;

  k_topk<<<(NB * NLOW + 255) / 256, 256, 0, stream>>>(cmap, idx);
  k_proj<<<NB * NSEQ / BM, 256, 0, stream>>>(qf, Wq, bq, qhp);
  k_proj<<<NB * NSEQ / BM, 256, 0, stream>>>(kf, Wk, bk, khp);
  k_proj<<<NB * NSEQ / BM, 256, 0, stream>>>(vf, Wv, bv, vhp);
  k_attn<<<NB * NLOW, 256, 0, stream>>>(qhp, khp, vhp, idx, xh);
  k_outproj<<<NB * NSEQ / BM, 256, 0, stream>>>(xh, Wo, bo, out);
}
